// MAMBADecoder_72834055405756
// MI455X (gfx1250) — matrix operand order repaired and checked by reading, not compiled, not run on hardware
//
#include <hip/hip_runtime.h>
#include <hip/hip_bf16.h>
#include <math.h>

// ---------------------------------------------------------------------------
// Mamba decoder for MI455X (gfx1250, wave32).
// GEMMs: v_wmma_f32_16x16x32_bf16; f32->bf16 staged through LDS with
// float4 global loads, packed cvt, vector LDS stores; global_prefetch ahead.
// Block tile 128x128 (8 waves, wave tile 32x64, 2x4 accumulators),
// KT=64 -> 16 WMMAs per K-iteration.
// B=4, L=512, D=768, NL=12, DI=1536, N=16, K=4, R=48;  M = B*L = 2048.
// ---------------------------------------------------------------------------

#define BB 4
#define LL 512
#define DD 768
#define NLAYER 12
#define DIM_I 1536
#define NSTATE 16
#define KCONV 4
#define RRANK 48
#define KPAD 64           // dt-proj K padded 48 -> 64
#define MROWS (BB * LL)   // 2048

typedef __attribute__((ext_vector_type(16))) __bf16 v16bf;
typedef __attribute__((ext_vector_type(8)))  __bf16 bf16x8;
typedef __attribute__((ext_vector_type(8)))  float  v8f;

// ---------------- Embedding: wte[ids] + wpe[l] -----------------------------
__global__ __launch_bounds__(256) void embed_kernel(
    const int* __restrict__ ids, const float* __restrict__ wte,
    const float* __restrict__ wpe, float* __restrict__ out) {
  int row = blockIdx.x;
  int l = row % LL;
  int tok = ids[row];
  #pragma unroll
  for (int j = 0; j < 3; ++j) {
    int d = j * 256 + threadIdx.x;
    out[(size_t)row * DD + d] = wte[(size_t)tok * DD + d] + wpe[(size_t)l * DD + d];
  }
}

// ---------------- Residual-add + LayerNorm ---------------------------------
__global__ __launch_bounds__(256) void add_ln_kernel(
    float* __restrict__ resid, const float* __restrict__ hid,
    const float* __restrict__ g, const float* __restrict__ b,
    float* __restrict__ out, int first) {
  int row = blockIdx.x;
  int tid = threadIdx.x;
  int lane = tid & 31, wid = tid >> 5;
  __shared__ float red[8];
  __shared__ float red2[8];

  float vals[3];
  float s = 0.f;
  #pragma unroll
  for (int j = 0; j < 3; ++j) {
    int d = j * 256 + tid;
    size_t idx = (size_t)row * DD + d;
    float v;
    if (hid) {
      v = first ? hid[idx] : (resid[idx] + hid[idx]);
      resid[idx] = v;
    } else {
      v = resid[idx];
    }
    vals[j] = v;
    s += v;
  }
  #pragma unroll
  for (int o = 16; o > 0; o >>= 1) s += __shfl_xor(s, o, 32);
  if (lane == 0) red[wid] = s;
  __syncthreads();
  if (tid == 0) { float t = 0.f; for (int i = 0; i < 8; ++i) t += red[i]; red[0] = t; }
  __syncthreads();
  float mean = red[0] * (1.0f / DD);

  float vs = 0.f;
  #pragma unroll
  for (int j = 0; j < 3; ++j) { float dv = vals[j] - mean; vs += dv * dv; }
  #pragma unroll
  for (int o = 16; o > 0; o >>= 1) vs += __shfl_xor(vs, o, 32);
  if (lane == 0) red2[wid] = vs;
  __syncthreads();
  if (tid == 0) { float t = 0.f; for (int i = 0; i < 8; ++i) t += red2[i]; red2[0] = t; }
  __syncthreads();
  float rstd = rsqrtf(red2[0] * (1.0f / DD) + 1e-5f);

  #pragma unroll
  for (int j = 0; j < 3; ++j) {
    int d = j * 256 + tid;
    out[(size_t)row * DD + d] = (vals[j] - mean) * rstd * g[d] + b[d];
  }
}

// ---------------- WMMA GEMM ------------------------------------------------
// Out[M x Nout] = act(X[M x Kd] * W[Nout x Kd]^T + bias)
// Requires: M % 128 == 0, Kd % 64 == 0, all row strides % 16 floats == 0.
#define MT 128
#define NT 128
#define KT 64
#define LDSK (KT + 8)     // 72 elems = 144 B, 16B-aligned stride

__device__ __forceinline__ bf16x8 pack_bf8(float4 a, float4 b) {
  bf16x8 p;
  p[0] = (__bf16)a.x; p[1] = (__bf16)a.y; p[2] = (__bf16)a.z; p[3] = (__bf16)a.w;
  p[4] = (__bf16)b.x; p[5] = (__bf16)b.y; p[6] = (__bf16)b.z; p[7] = (__bf16)b.w;
  return p;
}

__global__ __launch_bounds__(256) void gemm_wmma_kernel(
    const float* __restrict__ X, int ldx,
    const float* __restrict__ W, int Kd, int Nout,
    float* __restrict__ Out, int ldo,
    const float* __restrict__ bias, int act) {
  __shared__ __bf16 Asm[MT][LDSK];
  __shared__ __bf16 Bsm[NT][LDSK];

  const int tid  = threadIdx.x;
  const int lane = tid & 31;
  const int wave = tid >> 5;
  const int wm = wave >> 1;            // 0..3  (32 rows each)
  const int wn = wave & 1;             // 0..1  (64 cols each)
  const int blockM = blockIdx.x * MT;
  const int blockN = blockIdx.y * NT;

  v8f acc[2][4];
  #pragma unroll
  for (int mi = 0; mi < 2; ++mi)
    #pragma unroll
    for (int ni = 0; ni < 4; ++ni)
      #pragma unroll
      for (int r = 0; r < 8; ++r) acc[mi][ni][r] = 0.f;

  // A and B staging: 128 rows x 64 cols each, 32 elems/thread (8x float4)
  const int st_row = tid >> 1;
  const int st_c0  = (tid & 1) * 32;
  const int gnB    = blockN + st_row;
  const bool bok   = (gnB < Nout);

  const float* aptr = X + (size_t)(blockM + st_row) * ldx + st_c0;
  const float* bptr = W + (size_t)gnB * Kd + st_c0;

  for (int kc = 0; kc < Kd; kc += KT) {
    if (kc + KT < Kd) {
      __builtin_prefetch(aptr + kc + KT, 0, 1);
      __builtin_prefetch(bptr + kc + KT, 0, 1);
    }
    // ---- stage A (unconditional: M and Kd are tile multiples) ----
    {
      const float4* ap = (const float4*)(aptr + kc);
      #pragma unroll
      for (int q = 0; q < 4; ++q) {
        float4 f0 = ap[2 * q], f1 = ap[2 * q + 1];
        *(bf16x8*)&Asm[st_row][st_c0 + q * 8] = pack_bf8(f0, f1);
      }
    }
    // ---- stage B (single row-level guard for ragged Nout) ----
    if (bok) {
      const float4* bp = (const float4*)(bptr + kc);
      #pragma unroll
      for (int q = 0; q < 4; ++q) {
        float4 f0 = bp[2 * q], f1 = bp[2 * q + 1];
        *(bf16x8*)&Bsm[st_row][st_c0 + q * 8] = pack_bf8(f0, f1);
      }
    } else {
      bf16x8 z;
      #pragma unroll
      for (int e = 0; e < 8; ++e) z[e] = (__bf16)0.f;
      #pragma unroll
      for (int q = 0; q < 4; ++q)
        *(bf16x8*)&Bsm[st_row][st_c0 + q * 8] = z;
    }
    __syncthreads();

    // ---- two 32-deep WMMA passes over the staged 64-deep tile ----
    const int half = lane >> 4;
    const int l16  = lane & 15;
    #pragma unroll
    for (int kk = 0; kk < 2; ++kk) {
      const int ko = kk * 32;
      v16bf afrag[2], bfrag[4];
      #pragma unroll
      for (int mi = 0; mi < 2; ++mi) {
        int arow = wm * 32 + mi * 16 + l16;
        bf16x8 lo = *(const bf16x8*)&Asm[arow][ko + half * 8];
        bf16x8 hi = *(const bf16x8*)&Asm[arow][ko + 16 + half * 8];
        afrag[mi] = __builtin_shufflevector(lo, hi, 0, 1, 2, 3, 4, 5, 6, 7,
                                            8, 9, 10, 11, 12, 13, 14, 15);
      }
      #pragma unroll
      for (int ni = 0; ni < 4; ++ni) {
        int bcol = wn * 64 + ni * 16 + l16;
        bf16x8 lo = *(const bf16x8*)&Bsm[bcol][ko + half * 8];
        bf16x8 hi = *(const bf16x8*)&Bsm[bcol][ko + 16 + half * 8];
        bfrag[ni] = __builtin_shufflevector(lo, hi, 0, 1, 2, 3, 4, 5, 6, 7,
                                            8, 9, 10, 11, 12, 13, 14, 15);
      }
      #pragma unroll
      for (int mi = 0; mi < 2; ++mi)
        #pragma unroll
        for (int ni = 0; ni < 4; ++ni)
          acc[mi][ni] = __builtin_amdgcn_wmma_f32_16x16x32_bf16(
              false, afrag[mi], false, bfrag[ni],
              (short)0, acc[mi][ni], false, false);
    }
    __syncthreads();
  }

  // ---- epilogue: C layout lane l, vgpr r -> row=(l<16?r:r+8), col=l%16 ----
  const int l16 = lane & 15;
  const int rofs = (lane < 16) ? 0 : 8;
  #pragma unroll
  for (int ni = 0; ni < 4; ++ni) {
    int col = blockN + wn * 64 + ni * 16 + l16;
    bool cok = (col < Nout);
    float bv = (bias && cok) ? bias[col] : 0.f;
    #pragma unroll
    for (int mi = 0; mi < 2; ++mi) {
      #pragma unroll
      for (int r = 0; r < 8; ++r) {
        if (cok) {
          int row = blockM + wm * 32 + mi * 16 + r + rofs;
          float v = acc[mi][ni][r] + bv;
          if (act == 1) v = (v > 20.f) ? v : log1pf(__expf(v));  // softplus
          Out[(size_t)row * ldo + col] = v;
        }
      }
    }
  }
}

// ---------------- Depthwise causal conv (K=4) + SiLU -----------------------
__global__ __launch_bounds__(256) void conv_silu_kernel(
    const float* __restrict__ xz, const float* __restrict__ cW,
    const float* __restrict__ cb, float* __restrict__ xi) {
  int idx = blockIdx.x * 256 + threadIdx.x;   // over B*L*DI
  int c = idx % DIM_I;
  int l = (idx / DIM_I) % LL;
  int b = idx / (DIM_I * LL);
  float accv = cb[c];
  #pragma unroll
  for (int j = 0; j < KCONV; ++j) {
    int ls = l - (KCONV - 1) + j;
    if (ls >= 0)
      accv += cW[c * KCONV + j] * xz[((size_t)(b * LL + ls)) * (2 * DIM_I) + c];
  }
  xi[idx] = accv / (1.f + __expf(-accv));
}

// ---------------- K-padding copies for dt-proj -----------------------------
__global__ __launch_bounds__(256) void pad_dt_kernel(
    const float* __restrict__ dbl, float* __restrict__ dtpad) {
  int idx = blockIdx.x * 256 + threadIdx.x;   // over 2048*64
  int k = idx & (KPAD - 1);
  int row = idx >> 6;
  dtpad[idx] = (k < RRANK) ? dbl[row * 80 + k] : 0.f;
}
__global__ __launch_bounds__(256) void pad_dtw_kernel(
    const float* __restrict__ dtW, float* __restrict__ wpad) {
  int idx = blockIdx.x * 256 + threadIdx.x;   // over 1536*64
  int k = idx & (KPAD - 1);
  int n = idx >> 6;
  wpad[idx] = (k < RRANK) ? dtW[n * RRANK + k] : 0.f;
}

// ---------------- Selective scan -------------------------------------------
__global__ __launch_bounds__(256) void scan_kernel(
    const float* __restrict__ xi, const float* __restrict__ delta,
    const float* __restrict__ dblb, const float* __restrict__ a_log,
    float* __restrict__ yscan) {
  const int n  = threadIdx.x & 15;
  const int dl = threadIdx.x >> 4;
  const int d  = blockIdx.x * 16 + dl;
  const int b  = blockIdx.y;
  const float a = -__expf(a_log[d * NSTATE + n]);
  float h = 0.f;
  for (int l = 0; l < LL; ++l) {
    size_t base = (size_t)(b * LL + l);
    float u  = xi[base * DIM_I + d];
    float dt = delta[base * DIM_I + d];
    float bm = dblb[base * 80 + RRANK + n];
    float cm = dblb[base * 80 + RRANK + NSTATE + n];
    h = h * __expf(dt * a) + dt * u * bm;
    float p = h * cm;
    p += __shfl_xor(p, 1, 32);
    p += __shfl_xor(p, 2, 32);
    p += __shfl_xor(p, 4, 32);
    p += __shfl_xor(p, 8, 32);
    if (n == 0) yscan[base * DIM_I + d] = p;
  }
}

// ---------------- Gate: y = (yscan + xi*Dres) * silu(z) --------------------
__global__ __launch_bounds__(256) void gate_kernel(
    float* __restrict__ yscan, const float* __restrict__ xi,
    const float* __restrict__ dres, const float* __restrict__ xz) {
  int idx = blockIdx.x * 256 + threadIdx.x;   // over B*L*DI
  int c = idx % DIM_I;
  int row = idx / DIM_I;
  float z = xz[(size_t)row * (2 * DIM_I) + DIM_I + c];
  float y = yscan[idx] + xi[idx] * dres[c];
  yscan[idx] = y * (z / (1.f + __expf(-z)));
}

// ---------------------------------------------------------------------------
extern "C" void kernel_launch(void* const* d_in, const int* in_sizes, int n_in,
                              void* d_out, int out_size, void* d_ws, size_t ws_size,
                              hipStream_t stream) {
  const int*   ids   = (const int*)  d_in[0];
  const float* wte   = (const float*)d_in[1];
  const float* wpe   = (const float*)d_in[2];
  const float* ln_g  = (const float*)d_in[3];
  const float* ln_b  = (const float*)d_in[4];
  const float* in_W  = (const float*)d_in[5];
  const float* convW = (const float*)d_in[6];
  const float* convB = (const float*)d_in[7];
  const float* xp_W  = (const float*)d_in[8];
  const float* dt_W  = (const float*)d_in[9];
  const float* dt_b  = (const float*)d_in[10];
  const float* a_log = (const float*)d_in[11];
  const float* dres  = (const float*)d_in[12];
  const float* out_W = (const float*)d_in[13];
  const float* nf_g  = (const float*)d_in[14];
  const float* nf_b  = (const float*)d_in[15];
  const float* fin_g = (const float*)d_in[16];
  const float* fin_b = (const float*)d_in[17];

  // Workspace layout (floats); every buffer offset is a multiple of 16 floats.
  float* ws = (float*)d_ws;
  const size_t RD = (size_t)MROWS * DD;
  const size_t RI = (size_t)MROWS * DIM_I;
  float* hidden = ws;
  float* resid  = hidden + RD;
  float* hbuf   = resid + RD;
  float* xz     = hbuf + RD;                       // 2048*3072
  float* xibuf  = xz + (size_t)MROWS * 2 * DIM_I;  // 2048*1536
  float* dblb   = xibuf + RI;                      // 2048*80
  float* deltab = dblb + (size_t)MROWS * 80;       // 2048*1536
  float* yscan  = deltab + RI;                     // 2048*1536
  float* tmp    = yscan + RI;                      // 2048*768
  float* dtpad  = tmp + RD;                        // 2048*64
  float* wpad   = dtpad + (size_t)MROWS * KPAD;    // 1536*64

  const dim3 blk(256);
  const dim3 grid_rows(MROWS);
  const dim3 grid_ew((MROWS * DIM_I) / 256);
  const dim3 grid_scan(DIM_I / 16, BB);

  embed_kernel<<<grid_rows, blk, 0, stream>>>(ids, wte, wpe, hidden);

  for (int i = 0; i < NLAYER; ++i) {
    const float* lW_in  = in_W  + (size_t)i * 2 * DIM_I * DD;
    const float* lW_cv  = convW + (size_t)i * DIM_I * KCONV;
    const float* lB_cv  = convB + (size_t)i * DIM_I;
    const float* lW_xp  = xp_W  + (size_t)i * (RRANK + 2 * NSTATE) * DIM_I;
    const float* lW_dt  = dt_W  + (size_t)i * DIM_I * RRANK;
    const float* lB_dt  = dt_b  + (size_t)i * DIM_I;
    const float* lA     = a_log + (size_t)i * DIM_I * NSTATE;
    const float* lD     = dres  + (size_t)i * DIM_I;
    const float* lW_out = out_W + (size_t)i * DD * DIM_I;

    add_ln_kernel<<<grid_rows, blk, 0, stream>>>(
        resid, hidden, ln_g + (size_t)i * DD, ln_b + (size_t)i * DD, hbuf, i == 0);

    // in-proj: xz = hbuf @ in_W^T    (2048x768 x 768x3072)
    gemm_wmma_kernel<<<dim3(MROWS / MT, 2 * DIM_I / NT), blk, 0, stream>>>(
        hbuf, DD, lW_in, DD, 2 * DIM_I, xz, 2 * DIM_I, nullptr, 0);

    conv_silu_kernel<<<grid_ew, blk, 0, stream>>>(xz, lW_cv, lB_cv, xibuf);

    // xp-proj: dbl = xi @ xp_W^T     (2048x1536 x 1536x80), one ragged N-block
    gemm_wmma_kernel<<<dim3(MROWS / MT, (RRANK + 2 * NSTATE + NT - 1) / NT), blk, 0, stream>>>(
        xibuf, DIM_I, lW_xp, DIM_I, RRANK + 2 * NSTATE, dblb, 80, nullptr, 0);

    // K-pad dt activations (48->64) and dt weights, then
    // dt-proj + softplus: delta = softplus(dtpad @ wpad^T + dt_b)
    pad_dt_kernel<<<dim3((MROWS * KPAD) / 256), blk, 0, stream>>>(dblb, dtpad);
    pad_dtw_kernel<<<dim3((DIM_I * KPAD) / 256), blk, 0, stream>>>(lW_dt, wpad);
    gemm_wmma_kernel<<<dim3(MROWS / MT, DIM_I / NT), blk, 0, stream>>>(
        dtpad, KPAD, wpad, KPAD, DIM_I, deltab, DIM_I, lB_dt, 1);

    scan_kernel<<<grid_scan, blk, 0, stream>>>(xibuf, deltab, dblb, lA, yscan);

    gate_kernel<<<grid_ew, blk, 0, stream>>>(yscan, xibuf, lD, xz);

    // out-proj: hidden = yscan @ out_W^T   (2048x1536 x 1536x768)
    gemm_wmma_kernel<<<dim3(MROWS / MT, DD / NT), blk, 0, stream>>>(
        yscan, DIM_I, lW_out, DIM_I, DD, hidden, DD, nullptr, 0);
  }

  add_ln_kernel<<<grid_rows, blk, 0, stream>>>(resid, hidden, nf_g, nf_b, tmp, 0);
  add_ln_kernel<<<grid_rows, blk, 0, stream>>>(tmp, nullptr, fin_g, fin_b, (float*)d_out, 0);
}
